// MHA_8727373545917
// MI455X (gfx1250) — matrix operand order repaired and checked by reading, not compiled, not run on hardware
//
#include <hip/hip_runtime.h>
#include <hip/hip_bf16.h>

// ---- WMMA vector types (gfx1250, wave32) ----
typedef __attribute__((ext_vector_type(16))) __bf16 v16bf;
typedef __attribute__((ext_vector_type(8)))  __bf16 v8bf;
typedef __attribute__((ext_vector_type(8)))  float  v8f;

#define BDIM 32
#define TDIM 512
#define CDIM 256
#define NH   8
#define HDIM 32
#define MTOT (BDIM * TDIM)      // 16384 rows
#define QK_SCALE 0.17677669529663688f   // 1/sqrt(32)

// ---- fp32 -> bf16: use the hardware conversion (RTNE) ----
__device__ __forceinline__ __bf16 f2bf(float f) {
    return (__bf16)f;
}

__device__ __forceinline__ v16bf bfcat(v8bf lo, v8bf hi) {
    v16bf r;
#pragma unroll
    for (int i = 0; i < 8; ++i) { r[i] = lo[i]; r[8 + i] = hi[i]; }
    return r;
}

// A-operand 16x32 bf16 from an fp32 row (row-major, K contiguous).
// Layout: lanes 0-15 hold K 0..15; lanes 16-31 hold K 16..31.
__device__ __forceinline__ v16bf a_from_f32(const float* row, int kk, int lane) {
    int base = kk + ((lane & 16) ? 16 : 0);
    float4 a0 = *(const float4*)(row + base);
    float4 a1 = *(const float4*)(row + base + 4);
    float4 b0 = *(const float4*)(row + base + 8);
    float4 b1 = *(const float4*)(row + base + 12);
    v16bf r;
    r[0] = f2bf(a0.x); r[1] = f2bf(a0.y); r[2]  = f2bf(a0.z); r[3]  = f2bf(a0.w);
    r[4] = f2bf(a1.x); r[5] = f2bf(a1.y); r[6]  = f2bf(a1.z); r[7]  = f2bf(a1.w);
    r[8] = f2bf(b0.x); r[9] = f2bf(b0.y); r[10] = f2bf(b0.z); r[11] = f2bf(b0.w);
    r[12]= f2bf(b1.x); r[13]= f2bf(b1.y); r[14] = f2bf(b1.z); r[15] = f2bf(b1.w);
    return r;
}

// A-operand 16x32 bf16 from a bf16 row (row-major, K contiguous).
__device__ __forceinline__ v16bf a_from_bf16(const __bf16* row, int kk, int lane) {
    int base = kk + ((lane & 16) ? 16 : 0);
    v8bf lo = *(const v8bf*)(row + base);
    v8bf hi = *(const v8bf*)(row + base + 8);
    return bfcat(lo, hi);
}

#define WMMA_BF16(A, B, C) \
    __builtin_amdgcn_wmma_f32_16x16x32_bf16(false, (A), false, (B), (short)0, (C), false, false)

// ---------------------------------------------------------------------------
// Kernel 1: weight transpose + bf16 convert.  Wt[n*K + k] = bf16(W[k*N + n])
// ---------------------------------------------------------------------------
__global__ void mha_wtrans(const float* __restrict__ W, __bf16* __restrict__ Wt,
                           int K, int N) {
    int idx = blockIdx.x * blockDim.x + threadIdx.x;
    if (idx >= K * N) return;
    int k = idx / N, n = idx % N;
    Wt[(size_t)n * K + k] = f2bf(W[(size_t)k * N + n]);
}

// ---------------------------------------------------------------------------
// Kernel 2: QKV GEMM.  One wave computes a 16x64 tile of qkv = x @ Wqkv + b.
// Routes columns into Q [B,H,T,D] (pre-scaled), K [B,H,T,D], V^T [B,H,D,T].
// ---------------------------------------------------------------------------
__global__ void mha_qkv_gemm(const float* __restrict__ x,
                             const __bf16* __restrict__ Wt,   // [768, 256] bf16
                             const float* __restrict__ bias,  // [768]
                             __bf16* __restrict__ Qb,
                             __bf16* __restrict__ Kb,
                             __bf16* __restrict__ Vt) {
    int lane = threadIdx.x & 31;
    int wg   = blockIdx.x * (blockDim.x >> 5) + (threadIdx.x >> 5);
    int mt = wg / 12;            // 1024 row tiles
    int ng = wg % 12;            // 12 groups of 4 column tiles
    int m0 = mt * 16;
    int n0 = ng * 64;

    const float* arow = x + (size_t)(m0 + (lane & 15)) * CDIM;
    int koff = (lane & 16) ? 16 : 0;

    v8f acc[4] = {v8f{}, v8f{}, v8f{}, v8f{}};
    const __bf16* brow[4];
#pragma unroll
    for (int j = 0; j < 4; ++j)
        brow[j] = Wt + (size_t)(n0 + j * 16 + (lane & 15)) * CDIM + koff;

#pragma unroll
    for (int kk = 0; kk < CDIM; kk += 32) {
        v16bf A = a_from_f32(arow, kk, lane);
#pragma unroll
        for (int j = 0; j < 4; ++j) {
            v16bf B = *(const v16bf*)(brow[j] + kk);
            acc[j] = WMMA_BF16(A, B, acc[j]);
        }
    }

    int half8 = (lane & 16) ? 8 : 0;
#pragma unroll
    for (int j = 0; j < 4; ++j) {
        int ncol = n0 + j * 16 + (lane & 15);
        float bv = bias[ncol];
        int sel = ncol >> 8;         // 0=Q, 1=K, 2=V
        int c   = ncol & 255;
        int h   = c >> 5;
        int d   = c & 31;
#pragma unroll
        for (int r = 0; r < 8; ++r) {
            int M = m0 + r + half8;
            int b = M >> 9;          // / 512
            int t = M & 511;
            float v = acc[j][r] + bv;
            size_t bh = (size_t)(b * NH + h);
            if (sel == 0) {
                Qb[(bh * TDIM + t) * HDIM + d] = f2bf(v * QK_SCALE);
            } else if (sel == 1) {
                Kb[(bh * TDIM + t) * HDIM + d] = f2bf(v);
            } else {
                Vt[(bh * HDIM + d) * TDIM + t] = f2bf(v);
            }
        }
    }
}

// ---------------------------------------------------------------------------
// Kernel 3: causal flash attention.  One wave per (b, h, 16-query tile).
// Online softmax over 32-key chunks; S and O via WMMA; P goes D-layout ->
// A-layout through a per-wave LDS slab.  Row sums are kept as per-lane
// partials (alpha is row-uniform) and reduced once at the end.
// ---------------------------------------------------------------------------
__global__ void mha_flash_attn(const __bf16* __restrict__ Qb,
                               const __bf16* __restrict__ Kb,
                               const __bf16* __restrict__ Vt,
                               __bf16* __restrict__ AO) {   // [B,T,C] bf16
    __shared__ __bf16 Pshm[8 * 16 * 32];   // 8 waves/block * (16x32) tile
    int lane  = threadIdx.x & 31;
    int wslab = threadIdx.x >> 5;
    int wg    = blockIdx.x * (blockDim.x >> 5) + wslab;
    int qt = wg & 31;                      // 32 query tiles per head
    int bh = wg >> 5;                      // 0..255
    int b = bh >> 3, h = bh & 7;
    int q0 = qt * 16;
    __bf16* P = Pshm + wslab * 512;

    int col   = lane & 15;
    int half8 = (lane & 16) ? 8 : 0;
    int dg    = (lane & 16) ? 16 : 0;      // B-operand K-half select

    // Q A-operand, loaded once (row = q0 + lane%16, K = head dim).
    const __bf16* qrow = Qb + ((size_t)bh * TDIM + q0 + col) * HDIM;
    v16bf Aq = a_from_bf16(qrow, 0, lane);

    const __bf16* kbase_ptr = Kb + ((size_t)bh * TDIM + col) * HDIM + dg;
    const __bf16* v0_ptr = Vt + ((size_t)bh * HDIM + col) * TDIM + dg;
    const __bf16* v1_ptr = Vt + ((size_t)bh * HDIM + 16 + col) * TDIM + dg;

    float mrow[8], lrow[8];                // lrow = per-lane partial row sum
    v8f O0 = {}, O1 = {};
#pragma unroll
    for (int r = 0; r < 8; ++r) { mrow[r] = -1e30f; lrow[r] = 0.f; }

    int kcmax = qt >> 1;
    for (int kc = 0; kc <= kcmax; ++kc) {
        int kbase = kc * 32;
        int key0  = kbase + col;

        // S = Q @ K^T  (two 16-key sub-tiles)
        v16bf B0 = *(const v16bf*)(kbase_ptr + (size_t)kbase * HDIM);
        v16bf B1 = *(const v16bf*)(kbase_ptr + (size_t)(kbase + 16) * HDIM);
        v8f Z = {};
        v8f S0 = WMMA_BF16(Aq, B0, Z);
        v8f S1 = WMMA_BF16(Aq, B1, Z);

        // Prefetch next chunk's K rows into cache while we do softmax VALU.
        if (kc < kcmax) {
            __builtin_prefetch(kbase_ptr + (size_t)(kbase + 32) * HDIM, 0, 3);
            __builtin_prefetch(v0_ptr + kbase + 32, 0, 3);
            __builtin_prefetch(v1_ptr + kbase + 32, 0, 3);
        }

        // Online softmax per query row (rows live per lane-half; reduce max
        // over the 16 lanes of the half; sums stay per-lane partial).
#pragma unroll
        for (int r = 0; r < 8; ++r) {
            int qi = q0 + r + half8;
            bool v0 = key0 <= qi;
            bool v1 = (key0 + 16) <= qi;
            float s0 = v0 ? S0[r] : -1e30f;
            float s1 = v1 ? S1[r] : -1e30f;
            float tmax = fmaxf(s0, s1);
#pragma unroll
            for (int off = 1; off < 16; off <<= 1)
                tmax = fmaxf(tmax, __shfl_xor(tmax, off, 32));
            float mnew  = fmaxf(mrow[r], tmax);
            float alpha = __expf(mrow[r] - mnew);    // row-uniform across half
            float p0 = v0 ? __expf(S0[r] - mnew) : 0.f;
            float p1 = v1 ? __expf(S1[r] - mnew) : 0.f;
            lrow[r] = lrow[r] * alpha + (p0 + p1);   // per-lane partial
            mrow[r] = mnew;
            O0[r] *= alpha;
            O1[r] *= alpha;
            int mloc = r + half8;
            P[mloc * 32 + col]      = f2bf(p0);      // D-layout scatter to LDS
            P[mloc * 32 + col + 16] = f2bf(p1);
        }

        // Fence LDS stores before re-reading in A layout (same wave, cross-lane).
        asm volatile("s_wait_dscnt 0x0" ::: "memory");

        v16bf Pa = a_from_bf16(P + (size_t)col * 32, 0, lane);

        // O += P @ V  (V^T rows are contiguous in key)
        v16bf V0 = *(const v16bf*)(v0_ptr + kbase);
        v16bf V1 = *(const v16bf*)(v1_ptr + kbase);
        O0 = WMMA_BF16(Pa, V0, O0);
        O1 = WMMA_BF16(Pa, V1, O1);
    }

    // Reduce per-lane partial row sums across the 16 lanes of each half,
    // then normalize and write attn output as [B,T,H*D] bf16.
#pragma unroll
    for (int r = 0; r < 8; ++r) {
        float rs = lrow[r];
#pragma unroll
        for (int off = 1; off < 16; off <<= 1)
            rs += __shfl_xor(rs, off, 32);
        float inv = 1.0f / rs;
        int t = q0 + r + half8;
        size_t base = ((size_t)b * TDIM + t) * CDIM + h * HDIM;
        AO[base + col]      = f2bf(O0[r] * inv);
        AO[base + col + 16] = f2bf(O1[r] * inv);
    }
}

// ---------------------------------------------------------------------------
// Kernel 4: output projection.  One wave = 16x64 tile of out = attn @ Wp + b.
// ---------------------------------------------------------------------------
__global__ void mha_proj_gemm(const __bf16* __restrict__ Ain,   // [16384,256] bf16
                              const __bf16* __restrict__ Wt,    // [256,256] bf16
                              const float* __restrict__ bias,
                              float* __restrict__ out) {
    int lane = threadIdx.x & 31;
    int wg   = blockIdx.x * (blockDim.x >> 5) + (threadIdx.x >> 5);
    int mt = wg >> 2;            // 1024 row tiles
    int ng = wg & 3;             // 4 groups of 4 column tiles
    int m0 = mt * 16;
    int n0 = ng * 64;

    const __bf16* arow = Ain + (size_t)(m0 + (lane & 15)) * CDIM;
    int koff = (lane & 16) ? 16 : 0;

    v8f acc[4] = {v8f{}, v8f{}, v8f{}, v8f{}};
    const __bf16* brow[4];
#pragma unroll
    for (int j = 0; j < 4; ++j)
        brow[j] = Wt + (size_t)(n0 + j * 16 + (lane & 15)) * CDIM + koff;

#pragma unroll
    for (int kk = 0; kk < CDIM; kk += 32) {
        v16bf A = a_from_bf16(arow, kk, lane);
#pragma unroll
        for (int j = 0; j < 4; ++j) {
            v16bf B = *(const v16bf*)(brow[j] + kk);
            acc[j] = WMMA_BF16(A, B, acc[j]);
        }
    }

    int half8 = (lane & 16) ? 8 : 0;
#pragma unroll
    for (int j = 0; j < 4; ++j) {
        int ncol = n0 + j * 16 + (lane & 15);
        float bv = bias[ncol];
#pragma unroll
        for (int r = 0; r < 8; ++r) {
            int M = m0 + r + half8;
            out[(size_t)M * CDIM + ncol] = acc[j][r] + bv;
        }
    }
}

// ---------------------------------------------------------------------------
extern "C" void kernel_launch(void* const* d_in, const int* in_sizes, int n_in,
                              void* d_out, int out_size, void* d_ws, size_t ws_size,
                              hipStream_t stream) {
    const float* x     = (const float*)d_in[0];   // [32,512,256]
    const float* Wqkv  = (const float*)d_in[1];   // [256,768]
    const float* bqkv  = (const float*)d_in[2];   // [768]
    const float* Wproj = (const float*)d_in[3];   // [256,256]
    const float* bproj = (const float*)d_in[4];   // [256]
    float* out = (float*)d_out;

    char* ws = (char*)d_ws;
    auto alignup = [](size_t s) { return (s + 255) & ~(size_t)255; };
    __bf16* WtQKV = (__bf16*)ws; ws += alignup((size_t)CDIM * 3 * CDIM * 2);       // 768x256
    __bf16* WtP   = (__bf16*)ws; ws += alignup((size_t)CDIM * CDIM * 2);           // 256x256
    __bf16* Qb    = (__bf16*)ws; ws += alignup((size_t)BDIM * NH * TDIM * HDIM * 2);
    __bf16* Kb    = (__bf16*)ws; ws += alignup((size_t)BDIM * NH * TDIM * HDIM * 2);
    __bf16* Vt    = (__bf16*)ws; ws += alignup((size_t)BDIM * NH * TDIM * HDIM * 2);
    __bf16* AO    = (__bf16*)ws; ws += alignup((size_t)MTOT * CDIM * 2);

    // 1. weight transposes (bf16 convert)
    mha_wtrans<<<(CDIM * 3 * CDIM + 255) / 256, 256, 0, stream>>>(Wqkv, WtQKV, CDIM, 3 * CDIM);
    mha_wtrans<<<(CDIM * CDIM + 255) / 256, 256, 0, stream>>>(Wproj, WtP, CDIM, CDIM);

    // 2. QKV GEMM: 1024 * 12 waves, 8 waves/block
    mha_qkv_gemm<<<1024 * 12 / 8, 256, 0, stream>>>(x, WtQKV, bqkv, Qb, Kb, Vt);

    // 3. flash attention: 256 heads * 32 q-tiles = 8192 waves
    mha_flash_attn<<<8192 / 8, 256, 0, stream>>>(Qb, Kb, Vt, AO);

    // 4. projection: 1024 * 4 waves
    mha_proj_gemm<<<1024 * 4 / 8, 256, 0, stream>>>(AO, WtP, bproj, out);
}
